// GSVR_3564822856016
// MI455X (gfx1250) — compile-verified, matrix operand order checked by reading, not run on hardware
//
#include <hip/hip_runtime.h>
#include <hip/hip_bf16.h>

typedef __attribute__((ext_vector_type(16))) _Float16 v16h;
typedef __attribute__((ext_vector_type(8)))  float    v8f;

#define GSVR_DELTA 0.05f
#define GSVR_EPS   1e-8f
// 0.5 * log2(e): exp(-0.5*x) == exp2(-HALF_LOG2E * x)
#define HALF_LOG2E 0.72134752044448170368f

// Packed per-gaussian record: 48 bytes, 16-aligned.
// [0..3]  mu.x mu.y mu.z color
// [4..7]  s00 s01 s02 s11
// [8..9]  s12 s22   (+2 pad floats)
struct __align__(16) GPack {
    float mx, my, mz, col;
    float s00, s01, s02, s11;
    float s12, s22, pad0, pad1;
};

// ---------------------------------------------------------------------------
// Pass 1: per-gaussian Sigma = R diag(s^2) R^T  (removes the 42x per-pair
// recompute redundancy in the reference), packed with mu + color.
// ---------------------------------------------------------------------------
__global__ void gsvr_precompute(const float* __restrict__ mu,
                                const float* __restrict__ scaling,
                                const float* __restrict__ rot,
                                const float* __restrict__ color,
                                float4* __restrict__ outp, int G)
{
    int g = blockIdx.x * blockDim.x + threadIdx.x;
    if (g >= G) return;

    float qw = rot[g * 4 + 0], qx = rot[g * 4 + 1];
    float qy = rot[g * 4 + 2], qz = rot[g * 4 + 3];
    float inv = __builtin_amdgcn_rsqf(qw * qw + qx * qx + qy * qy + qz * qz + GSVR_EPS);
    qw *= inv; qx *= inv; qy *= inv; qz *= inv;

    float x2 = qx * qx, y2 = qy * qy, z2 = qz * qz;
    float xy = qx * qy, xz = qx * qz, yz = qy * qz;
    float wx = qw * qx, wy = qw * qy, wz = qw * qz;

    float sx = scaling[g * 3 + 0], sy = scaling[g * 3 + 1], sz = scaling[g * 3 + 2];
    float m00 = (1.0f - 2.0f * (y2 + z2)) * sx, m01 = (2.0f * (xy - wz)) * sy, m02 = (2.0f * (xz + wy)) * sz;
    float m10 = (2.0f * (xy + wz)) * sx, m11 = (1.0f - 2.0f * (x2 + z2)) * sy, m12 = (2.0f * (yz - wx)) * sz;
    float m20 = (2.0f * (xz - wy)) * sx, m21 = (2.0f * (yz + wx)) * sy, m22 = (1.0f - 2.0f * (x2 + y2)) * sz;

    float s00 = m00 * m00 + m01 * m01 + m02 * m02;
    float s01 = m00 * m10 + m01 * m11 + m02 * m12;
    float s02 = m00 * m20 + m01 * m21 + m02 * m22;
    float s11 = m10 * m10 + m11 * m11 + m12 * m12;
    float s12 = m10 * m20 + m11 * m21 + m12 * m22;
    float s22 = m20 * m20 + m21 * m21 + m22 * m22;

    float4 a = make_float4(mu[g * 3 + 0], mu[g * 3 + 1], mu[g * 3 + 2], color[g]);
    float4 b = make_float4(s00, s01, s02, s11);
    float4 c = make_float4(s12, s22, 0.0f, 0.0f);
    outp[(size_t)g * 3 + 0] = a;
    outp[(size_t)g * 3 + 1] = b;
    outp[(size_t)g * 3 + 2] = c;
}

// Per-pair Mahalanobis weight (matches reference explicit 3x3 inverse; raw
// v_rcp_f32 and bare v_exp_f32: arg <= 0, underflow-to-zero is desired).
__device__ __forceinline__ float gsvr_weight(float vx, float vy, float vz,
                                             float a, float bb, float c,
                                             float ee, float f, float ii)
{
    float det = a * (ee * ii - f * f) - bb * (bb * ii - f * c) + c * (bb * f - ee * c);
    float invdet = __builtin_amdgcn_rcpf(det + GSVR_EPS);
    float i00 = (ee * ii - f * f) * invdet;
    float i01 = (c * f - bb * ii) * invdet;
    float i02 = (bb * f - c * ee) * invdet;
    float i11 = (a * ii - c * c) * invdet;
    float i12 = (c * bb - a * f) * invdet;
    float i22 = (a * ee - bb * bb) * invdet;
    float t0 = i00 * vx + i01 * vy + i02 * vz;
    float t1 = i01 * vx + i11 * vy + i12 * vz;
    float t2 = i02 * vx + i12 * vy + i22 * vz;
    float inner = vx * t0 + vy * t1 + vz * t2;
    return __builtin_amdgcn_exp2f(inner * -HALF_LOG2E);   // bare v_exp_f32
}

// ---------------------------------------------------------------------------
// Pass 2: one wave32 per tile of 16 points. Each lane computes 16 (point,k)
// weights positioned per the CDNA5 16-bit A-matrix 16x32 VGPR layout:
//   lane L (group0, L<16):  row M=L,    K in {0..7, 16..23}
//   lane L (group1, L>=16): row M=L-16, K in {8..15, 24..31}
//   element e of v16h -> K = kbase + (e<8 ? e : e+8), kbase = (L<16?0:8)
// Then TWO v_wmma_f32_16x16x32_f16 against an all-ones B reduce
// sum(w) and sum(w*color) over all K=32 for all 16 points at once.
// ---------------------------------------------------------------------------
__global__ void gsvr_wmma_kernel(const float* __restrict__ coords,
                                 const float* __restrict__ psf,
                                 const float4* __restrict__ gp,
                                 const int* __restrict__ topk,
                                 float* __restrict__ out, int N)
{
    int tid  = (int)(blockIdx.x * blockDim.x + threadIdx.x);
    int tile = tid >> 5;
    int lane = tid & 31;
    if (tile * 16 >= N) return;              // wave-uniform; grid sized exactly

    int prow = lane & 15;
    int p = tile * 16 + prow;

    // Per-point data (point shared by lanes L and L+16; both load, cache-hit).
    float cx = coords[p * 3 + 0];
    float cy = coords[p * 3 + 1];
    float cz = coords[p * 3 + 2];
    const float* Pp = psf + (size_t)p * 9;
    float p00 = Pp[0], p01 = Pp[1], p02 = Pp[2];
    float p11 = Pp[4], p12 = Pp[5], p22 = Pp[8];

    // Each lane's 16 top-k indices are two contiguous runs of 8:
    // [kbase .. kbase+7] and [kbase+16 .. kbase+23]  ->  4x global_load_b128.
    int kbase = (lane < 16) ? 0 : 8;
    const int* row = topk + (size_t)p * 32 + kbase;
    int4 i0 = *(const int4*)(row + 0);
    int4 i1 = *(const int4*)(row + 4);
    int4 i2 = *(const int4*)(row + 16);
    int4 i3 = *(const int4*)(row + 20);
    int gidx[16] = { i0.x, i0.y, i0.z, i0.w, i1.x, i1.y, i1.z, i1.w,
                     i2.x, i2.y, i2.z, i2.w, i3.x, i3.y, i3.z, i3.w };

    v16h va;   // w        (A operand 1)
    v16h vc;   // w*color  (A operand 2)
#pragma unroll
    for (int e = 0; e < 16; ++e) {
        int g = gidx[e];                          // element e <-> K per layout
        const float4* base = gp + (size_t)g * 3;
        float4 A = base[0];                       // mu.xyz, color
        float4 B = base[1];                       // s00 s01 s02 s11
        float2 C = *(const float2*)(base + 2);    // s12 s22

        float vx = cx - A.x, vy = cy - A.y, vz = cz - A.z;
        float w = gsvr_weight(vx, vy, vz,
                              B.x + p00, B.y + p01, B.z + p02,
                              B.w + p11, C.x + p12, C.y + p22);
        va[e] = (_Float16)w;
        vc[e] = (_Float16)(w * A.w);
    }

    v16h ones;
#pragma unroll
    for (int e = 0; e < 16; ++e) ones[e] = (_Float16)1.0f;

    v8f dw  = {};
    v8f dwc = {};
    // D[p][n] = sum_k A[p][k] for every column n (B is all ones)
    dw  = __builtin_amdgcn_wmma_f32_16x16x32_f16(false, va, false, ones,
                                                 (short)0, dw,  false, false);
    dwc = __builtin_amdgcn_wmma_f32_16x16x32_f16(false, vc, false, ones,
                                                 (short)0, dwc, false, false);

    // Recompute lane state post-WMMA (cheap; avoids a long live range that
    // previously spilled to scratch across the unrolled loop).
    int lane2 = (int)(__builtin_amdgcn_workitem_id_x() & 31);
    int r2 = lane2 & 15;

    // C/D layout: lane L<16 holds rows 0..7 in VGPRs 0..7 (col N=L),
    //             lane L>=16 holds rows 8..15. Lanes with r2<8 each emit one row.
    if (r2 < 8) {
        float sw = 0.0f, swc = 0.0f;
#pragma unroll
        for (int j = 0; j < 8; ++j) {
            if (r2 == j) { sw = dw[j]; swc = dwc[j]; }
        }
        int rowi = r2 + ((lane2 < 16) ? 0 : 8);
        out[tile * 16 + rowi] = swc * __builtin_amdgcn_rcpf(sw + GSVR_DELTA);
    }
}

// ---------------------------------------------------------------------------
// Fallback: fully fused scalar path (any K, no workspace needed).
// ---------------------------------------------------------------------------
__global__ void gsvr_scalar(const float* __restrict__ coords,
                            const float* __restrict__ psf,
                            const float* __restrict__ mu,
                            const float* __restrict__ scaling,
                            const float* __restrict__ rot,
                            const float* __restrict__ color,
                            const int* __restrict__ topk,
                            float* __restrict__ out, int N, int K)
{
    int n = blockIdx.x * blockDim.x + threadIdx.x;
    if (n >= N) return;
    float cx = coords[n * 3 + 0], cy = coords[n * 3 + 1], cz = coords[n * 3 + 2];
    const float* Pp = psf + (size_t)n * 9;
    float p00 = Pp[0], p01 = Pp[1], p02 = Pp[2];
    float p11 = Pp[4], p12 = Pp[5], p22 = Pp[8];
    float sumw = 0.0f, sumwc = 0.0f;
    for (int k = 0; k < K; ++k) {
        int g = topk[(size_t)n * K + k];
        float qw = rot[g * 4 + 0], qx = rot[g * 4 + 1];
        float qy = rot[g * 4 + 2], qz = rot[g * 4 + 3];
        float inv = __builtin_amdgcn_rsqf(qw * qw + qx * qx + qy * qy + qz * qz + GSVR_EPS);
        qw *= inv; qx *= inv; qy *= inv; qz *= inv;
        float x2 = qx * qx, y2 = qy * qy, z2 = qz * qz;
        float xy = qx * qy, xz = qx * qz, yz = qy * qz;
        float wx = qw * qx, wy = qw * qy, wz = qw * qz;
        float sx = scaling[g * 3 + 0], sy = scaling[g * 3 + 1], sz = scaling[g * 3 + 2];
        float m00 = (1.0f - 2.0f * (y2 + z2)) * sx, m01 = (2.0f * (xy - wz)) * sy, m02 = (2.0f * (xz + wy)) * sz;
        float m10 = (2.0f * (xy + wz)) * sx, m11 = (1.0f - 2.0f * (x2 + z2)) * sy, m12 = (2.0f * (yz - wx)) * sz;
        float m20 = (2.0f * (xz - wy)) * sx, m21 = (2.0f * (yz + wx)) * sy, m22 = (1.0f - 2.0f * (x2 + y2)) * sz;
        float s00 = m00 * m00 + m01 * m01 + m02 * m02;
        float s01 = m00 * m10 + m01 * m11 + m02 * m12;
        float s02 = m00 * m20 + m01 * m21 + m02 * m22;
        float s11 = m10 * m10 + m11 * m11 + m12 * m12;
        float s12 = m10 * m20 + m11 * m21 + m12 * m22;
        float s22 = m20 * m20 + m21 * m21 + m22 * m22;
        float vx = cx - mu[g * 3 + 0], vy = cy - mu[g * 3 + 1], vz = cz - mu[g * 3 + 2];
        float w = gsvr_weight(vx, vy, vz, s00 + p00, s01 + p01, s02 + p02,
                              s11 + p11, s12 + p12, s22 + p22);
        sumw += w;
        sumwc += w * color[g];
    }
    out[n] = sumwc * __builtin_amdgcn_rcpf(sumw + GSVR_DELTA);
}

extern "C" void kernel_launch(void* const* d_in, const int* in_sizes, int n_in,
                              void* d_out, int out_size, void* d_ws, size_t ws_size,
                              hipStream_t stream)
{
    const float* coords  = (const float*)d_in[0];
    const float* psf     = (const float*)d_in[1];
    const float* mu      = (const float*)d_in[2];
    const float* scaling = (const float*)d_in[3];
    const float* rot     = (const float*)d_in[4];
    const float* color   = (const float*)d_in[5];
    const int*   topk    = (const int*)d_in[6];
    float* out = (float*)d_out;

    int N = in_sizes[0] / 3;
    int G = in_sizes[2] / 3;
    int K = (N > 0) ? (in_sizes[6] / N) : 0;

    size_t need = (size_t)G * sizeof(GPack);
    if (K == 32 && (N % 16) == 0 && ws_size >= need) {
        float4* gp = (float4*)d_ws;
        gsvr_precompute<<<(G + 255) / 256, 256, 0, stream>>>(mu, scaling, rot, color, gp, G);
        int waves = N / 16;                 // one wave32 per 16-point tile
        long long threads = (long long)waves * 32;
        int blocks = (int)((threads + 255) / 256);
        gsvr_wmma_kernel<<<blocks, 256, 0, stream>>>(coords, psf, gp, topk, out, N);
    } else {
        gsvr_scalar<<<(N + 255) / 256, 256, 0, stream>>>(coords, psf, mu, scaling, rot,
                                                         color, topk, out, N, K);
    }
}
